// SVS_7773890806335
// MI455X (gfx1250) — matrix operand order repaired and checked by reading, not compiled, not run on hardware
//
#include <hip/hip_runtime.h>
#include <math.h>

#define Bsz   8
#define Tn    512
#define HOP   240
#define Ln    122880          // T*HOP
#define NMELS 80
#define Kh    80              // harmonics
#define NMAGH 256
#define NMAGN 80
#define NFFT  1024
#define SRf   24000.0f
#define INDIM 233
#define KPAD  256
#define HID   256
#define ROWS  4096            // B*T
#define CHUNK 1024
#define NCHNK 120             // L / CHUNK

typedef __attribute__((ext_vector_type(16))) _Float16 v16h;
typedef __attribute__((ext_vector_type(8)))  _Float16 h8;
typedef __attribute__((ext_vector_type(8)))  float    v8f;

// ---------------- device helpers ----------------
__device__ __forceinline__ float sigmoidf_(float x) { return 1.0f / (1.0f + expf(-x)); }
__device__ __forceinline__ float gelu_(float x) {
    float x3 = x * x * x;
    return 0.5f * x * (1.0f + tanhf(0.7978845608028654f * (x + 0.044715f * x3)));
}
__device__ __forceinline__ float scalefn_(float x) {
    // 2*sigmoid(x)^log(10) + 1e-7
    return 2.0f * powf(sigmoidf_(x), 2.302585092994046f) + 1e-7f;
}

// ---------------- weight packing into WMMA B-fragment layout ----------------
// packed index = (((ntile*8 + kk)*32 + lane)*16 + e)
// element e of lane: K = kk*32 + (lane>=16?16:0) + e ; col = ntile*16 + (lane&15)
__global__ void svs_pack_w(const float* __restrict__ W, _Float16* __restrict__ out,
                           int Korig, int N, int total) {
    int idx = blockIdx.x * 256 + threadIdx.x;
    if (idx >= total) return;
    int e     = idx & 15;
    int lane  = (idx >> 4) & 31;
    int kk    = (idx >> 9) & 7;
    int ntile = idx >> 12;
    int K   = kk * 32 + ((lane >> 4) << 4) + e;
    int col = ntile * 16 + (lane & 15);
    float v = (K < Korig) ? W[(size_t)K * N + col] : 0.0f;
    out[idx] = (_Float16)v;
}

// ---------------- feature build (concat + embeddings), f16, K padded to 256 ----
__global__ void svs_build_feat(const float* __restrict__ mel, const float* __restrict__ f0,
                               const int* __restrict__ seq, const int* __restrict__ sid,
                               const int* __restrict__ lid, const float* __restrict__ ptab,
                               const float* __restrict__ stab, const float* __restrict__ ltab,
                               _Float16* __restrict__ feat) {
    int idx = blockIdx.x * 256 + threadIdx.x;     // row*256 + col, total ROWS*256
    int col = idx & 255;
    int row = idx >> 8;
    int b = row >> 9;                             // T = 512
    float v;
    if (col < 80)        v = mel[(size_t)row * NMELS + col];
    else if (col == 80)  v = f0[row];
    else if (col < 209)  v = ptab[seq[row] * 128 + (col - 81)];
    else if (col < 225)  v = stab[sid[b] * 16 + (col - 209)];
    else if (col < 233)  v = ltab[lid[b] * 8 + (col - 225)];
    else                 v = 0.0f;
    feat[idx] = (_Float16)v;
}

// ---------------- WMMA GEMM: [4096,256]f16 x packed[256,N]f16 + bias, act ------
// one 16x16 tile per wave, 8 waves/block, K loop of 8 steps of 32
__global__ void __launch_bounds__(256) svs_wmma_gemm(
    const _Float16* __restrict__ A, const _Float16* __restrict__ Bp,
    const float* __restrict__ bias, float* __restrict__ outF32,
    _Float16* __restrict__ outF16, int N, int ntilesN, int act) {
    int lane = threadIdx.x & 31;
    int wave = threadIdx.x >> 5;
    int tile = blockIdx.x * 8 + wave;
    int mtile = tile / ntilesN;
    int ntile = tile % ntilesN;
    int mrow = mtile * 16 + (lane & 15);
    int aoff = (lane >> 4) * 16;
    const _Float16* aptr = A + (size_t)mrow * KPAD + aoff;
    const _Float16* bptr = Bp + ((size_t)ntile * 8 * 32 + lane) * 16;
    v8f acc = {};
    for (int kk = 0; kk < 8; ++kk) {
        v16h a, bfr;
        *((h8*)&a)     = *(const h8*)(aptr);        // K e0..7
        *((h8*)&a + 1) = *(const h8*)(aptr + 8);    // K e8..15
        bfr = *(const v16h*)(bptr);
        acc = __builtin_amdgcn_wmma_f32_16x16x32_f16(
            false, a, false, bfr, (short)0, acc, false, false);
        aptr += 32;
        bptr += 32 * 16;
    }
    int col = ntile * 16 + (lane & 15);
    float bv = bias[col];
    int rbase = mtile * 16 + ((lane >> 4) << 3);
    for (int r = 0; r < 8; ++r) {
        float v = acc[r] + bv;
        if (act == 1)      v = gelu_(v);
        else if (act == 2) v = scalefn_(v);
        size_t o = (size_t)(rbase + r) * N + col;
        if (outF32) outF32[o] = v;
        if (outF16) outF16[o] = (_Float16)v;
    }
}

// ---------------- elementwise scale_fn (src_param from harm_ctrl) -------------
__global__ void svs_scalefn(const float* __restrict__ in, float* __restrict__ out, int n) {
    int idx = blockIdx.x * 256 + threadIdx.x;
    if (idx < n) out[idx] = scalefn_(in[idx]);
}

// ---------------- expressive heads: 3 dots of 64 per row ----------------------
__global__ void svs_heads(const float* __restrict__ e,
                          const float* __restrict__ Wvr, const float* __restrict__ bvr,
                          const float* __restrict__ Wvd, const float* __restrict__ bvd,
                          const float* __restrict__ Wbr, const float* __restrict__ bbr,
                          float* __restrict__ vr, float* __restrict__ vd,
                          float* __restrict__ br) {
    __shared__ float wv[64], wd[64], wb[64];
    if (threadIdx.x < 64) {
        wv[threadIdx.x] = Wvr[threadIdx.x];
        wd[threadIdx.x] = Wvd[threadIdx.x];
        wb[threadIdx.x] = Wbr[threadIdx.x];
    }
    __syncthreads();
    int row = blockIdx.x * 256 + threadIdx.x;
    if (row >= ROWS) return;
    const float* ep = e + (size_t)row * 64;
    float a = bvr[0], d = bvd[0], c = bbr[0];
    for (int i = 0; i < 64; ++i) {
        float ev = ep[i];
        a += ev * wv[i]; d += ev * wd[i]; c += ev * wb[i];
    }
    vr[row] = 3.0f + 5.0f * sigmoidf_(a);
    vd[row] = 0.05f * sigmoidf_(d);
    br[row] = 0.1f * sigmoidf_(c);
}

// ---------------- f0 clamp + vibrato -----------------------------------------
__global__ void svs_f0vib(const float* __restrict__ f0, const float* __restrict__ vr,
                          const float* __restrict__ vd, float* __restrict__ f0v) {
    int idx = blockIdx.x * 256 + threadIdx.x;
    if (idx >= ROWS) return;
    int t = idx & 511;
    float f = fminf(fmaxf(f0[idx], 0.0f), 1000.0f);
    if (f < 80.0f) f = 1e-7f;
    float ph = 6.2831853071795864f * vr[idx] * ((float)t * 0.01f);
    f0v[idx] = f * (1.0f + vd[idx] * sinf(ph));
}

// ---------------- linear upsample frames -> samples ---------------------------
__global__ void svs_upsample(const float* __restrict__ f0v, float* __restrict__ f0up) {
    int idx = blockIdx.x * 256 + threadIdx.x;
    if (idx >= Bsz * Ln) return;
    int b = idx / Ln, l = idx % Ln;
    float pos = ((float)l + 0.5f) * (1.0f / 240.0f) - 0.5f;
    pos = fminf(fmaxf(pos, 0.0f), 511.0f);
    int i0 = (int)floorf(pos);
    int i1 = min(i0 + 1, 511);
    float w = pos - (float)i0;
    const float* fr = f0v + (size_t)b * Tn;
    f0up[idx] = fr[i0] * (1.0f - w) + fr[i1] * w;
}

// ---------------- phase cumsum: pass 1 (intra-chunk scan) ---------------------
__global__ void svs_scan1(const float* __restrict__ f0up, float* __restrict__ phasePart,
                          float* __restrict__ chunkSum) {
    __shared__ float ts[256];
    int row = blockIdx.x / NCHNK, chunk = blockIdx.x % NCHNK;
    const float* src = f0up + (size_t)row * Ln + (size_t)chunk * CHUNK;
    int base = threadIdx.x * 4;
    float v[4], s = 0.0f;
    for (int j = 0; j < 4; ++j) { v[j] = src[base + j] * (1.0f / SRf); s += v[j]; }
    ts[threadIdx.x] = s;
    __syncthreads();
    for (int off = 1; off < 256; off <<= 1) {
        float add = (threadIdx.x >= off) ? ts[threadIdx.x - off] : 0.0f;
        __syncthreads();
        ts[threadIdx.x] += add;
        __syncthreads();
    }
    float run = ts[threadIdx.x] - s;   // exclusive prefix of this thread
    float* dst = phasePart + (size_t)row * Ln + (size_t)chunk * CHUNK + base;
    for (int j = 0; j < 4; ++j) { run += v[j]; dst[j] = run; }
    if (threadIdx.x == 255) chunkSum[row * NCHNK + chunk] = ts[255];
}

// ---------------- phase cumsum: pass 2 (chunk-sum scan per row) ---------------
__global__ void svs_scan2(const float* __restrict__ chunkSum, float* __restrict__ chunkOff) {
    __shared__ float s[128];
    int row = blockIdx.x, i = threadIdx.x;
    float v = (i < NCHNK) ? chunkSum[row * NCHNK + i] : 0.0f;
    s[i] = v;
    __syncthreads();
    for (int off = 1; off < 128; off <<= 1) {
        float a = (i >= off) ? s[i - off] : 0.0f;
        __syncthreads();
        s[i] += a;
        __syncthreads();
    }
    if (i < NCHNK) chunkOff[row * NCHNK + i] = s[i] - v;   // exclusive
}

// ---------------- harmonic oscillator: Chebyshev sin(k*phi) recurrence --------
// amp reciprocals hoisted to LDS (kills 80 f32 divides per sample)
__global__ void svs_harm(const float* __restrict__ f0up, const float* __restrict__ phasePart,
                         const float* __restrict__ chunkOff, float* __restrict__ harm,
                         float coef) {
    __shared__ float invk[Kh];
    if (threadIdx.x < Kh) invk[threadIdx.x] = 1.0f / (float)(threadIdx.x + 1);
    __syncthreads();
    int idx = blockIdx.x * 256 + threadIdx.x;
    if (idx >= Bsz * Ln) return;
    int row = idx / Ln, i = idx % Ln;
    float ph = 6.2831853071795864f * (phasePart[idx] + chunkOff[row * NCHNK + (i >> 10)]);
    float f0u = f0up[idx];
    float s1, c;
    __sincosf(ph, &s1, &c);
    float twoC = 2.0f * c;
    float skm1 = 0.0f, sk = s1, sum = 0.0f;
#pragma unroll 8
    for (int k = 1; k <= Kh; ++k) {
        float amp = (f0u * (float)k < 12000.0f) ? invk[k - 1] : 0.0f;
        sum += amp * sk;
        float sn = twoC * sk - skm1;
        skm1 = sk; sk = sn;
    }
    harm[idx] = coef * sum;   // coef = 0.4 / H_80
}

// ---------------- per-frame zero-phase FFT filter (1024-pt radix-2 in LDS) ----
// shared twiddle table: tw[m] = (cos, sin)(+2*pi*m/1024); stage twiddle
// exp(sign*i*2*pi*j/len) = (tw[m].x, sign*tw[m].y) with m = j*(NFFT/len)
__device__ void fft_stages_(float2* X, const float2* tw, float sign) {
    int stride = NFFT / 2;
    for (int len = 2; len <= NFFT; len <<= 1, stride >>= 1) {
        int half = len >> 1;
        for (int idx = threadIdx.x; idx < NFFT / 2; idx += 256) {
            int blk = idx / half, j = idx % half;
            int pos = blk * len + j;
            float2 w = tw[j * stride];
            float cj = w.x, sj = sign * w.y;
            float2 u = X[pos], v = X[pos + half];
            float vr = v.x * cj - v.y * sj;
            float vi = v.x * sj + v.y * cj;
            X[pos]        = make_float2(u.x + vr, u.y + vi);
            X[pos + half] = make_float2(u.x - vr, u.y - vi);
        }
        __syncthreads();
    }
}
__device__ void bitrev_(float2* X) {
    for (int i = threadIdx.x; i < NFFT; i += 256) {
        int r = (int)(__brev((unsigned)i) >> 22);
        if (r > i) { float2 t = X[i]; X[i] = X[r]; X[r] = t; }
    }
}
__global__ void __launch_bounds__(256) svs_fft_filter(const float* __restrict__ audio,
                                                      const float* __restrict__ mags,
                                                      int n_mag, float* __restrict__ outp) {
    __shared__ float2 X[NFFT];
    __shared__ float2 tw[NFFT / 2];
    __shared__ float  magL[256];
    __shared__ float  g[513];
    int frame = blockIdx.x;                    // b*T + t
    int b = frame / Tn, t = frame % Tn;
    const float* src = audio + (size_t)b * Ln + (size_t)t * HOP;
    // twiddle table (built once, reused by forward and inverse FFT)
    for (int m = threadIdx.x; m < NFFT / 2; m += 256) {
        float sm, cm;
        __sincosf(6.2831853071795864f * (float)m * (1.0f / (float)NFFT), &sm, &cm);
        tw[m] = make_float2(cm, sm);
    }
    for (int i = threadIdx.x; i < n_mag; i += 256)
        magL[i] = mags[(size_t)frame * n_mag + i];
    __syncthreads();
    for (int k = threadIdx.x; k <= 512; k += 256) {
        float p = (float)k * (float)(n_mag - 1) * (1.0f / 512.0f);
        int i0 = (int)floorf(p);
        int i1 = min(i0 + 1, n_mag - 1);
        float w = p - (float)i0;
        g[k] = magL[i0] * (1.0f - w) + magL[i1] * w;
    }
    // load frame in bit-reversed order, zero-padded to 1024
    for (int i = threadIdx.x; i < NFFT; i += 256) {
        int r = (int)(__brev((unsigned)i) >> 22);
        float v = (i < HOP) ? src[i] : 0.0f;
        X[r] = make_float2(v, 0.0f);
    }
    __syncthreads();
    fft_stages_(X, tw, -1.0f);                 // forward DFT
    for (int k = threadIdx.x; k < NFFT; k += 256) {
        int kk = (k <= 512) ? k : NFFT - k;    // Hermitian-symmetric real gain
        float gv = g[kk];
        X[k].x *= gv; X[k].y *= gv;
    }
    __syncthreads();
    bitrev_(X);
    __syncthreads();
    fft_stages_(X, tw, 1.0f);                  // inverse DFT (unscaled)
    if (threadIdx.x < HOP)
        outp[(size_t)b * Ln + (size_t)t * HOP + threadIdx.x] =
            X[threadIdx.x].x * (1.0f / (float)NFFT);
}

// ---------------- final mix: out += repeat(breath,HOP) * noise ----------------
__global__ void svs_mix(float* __restrict__ out, const float* __restrict__ noiseF,
                        const float* __restrict__ breath) {
    int idx = blockIdx.x * 256 + threadIdx.x;
    if (idx >= Bsz * Ln) return;
    int b = idx / Ln, i = idx % Ln;
    out[idx] += breath[b * Tn + i / HOP] * noiseF[idx];
}

// ============================ host launcher ==================================
extern "C" void kernel_launch(void* const* d_in, const int* in_sizes, int n_in,
                              void* d_out, int out_size, void* d_ws, size_t ws_size,
                              hipStream_t stream) {
    (void)in_sizes; (void)n_in; (void)out_size; (void)ws_size;
    const float* mel   = (const float*)d_in[0];
    const float* f0    = (const float*)d_in[1];
    const float* noise = (const float*)d_in[2];
    const int*   seq   = (const int*)d_in[3];
    const int*   sid   = (const int*)d_in[4];
    const int*   lid   = (const int*)d_in[5];
    const float* ptab  = (const float*)d_in[6];
    const float* stab  = (const float*)d_in[7];
    const float* ltab  = (const float*)d_in[8];
    const float* W1 = (const float*)d_in[9],  *b1 = (const float*)d_in[10];
    const float* W2 = (const float*)d_in[11], *b2 = (const float*)d_in[12];
    const float* Wh = (const float*)d_in[13], *bh = (const float*)d_in[14];
    const float* Wn = (const float*)d_in[15], *bn = (const float*)d_in[16];
    const float* We = (const float*)d_in[17], *be = (const float*)d_in[18];
    const float* Wvr = (const float*)d_in[19], *bvr = (const float*)d_in[20];
    const float* Wvd = (const float*)d_in[21], *bvd = (const float*)d_in[22];
    const float* Wbr = (const float*)d_in[23], *bbr = (const float*)d_in[24];
    float* out = (float*)d_out;

    // bump allocator over workspace
    uintptr_t cur = (uintptr_t)d_ws;
    auto alloc = [&](size_t bytes) -> void* {
        cur = (cur + 255) & ~(uintptr_t)255;
        void* p = (void*)cur;
        cur += bytes;
        return p;
    };
    _Float16* w1p   = (_Float16*)alloc(65536 * 2);
    _Float16* w2p   = (_Float16*)alloc(65536 * 2);
    _Float16* whp   = (_Float16*)alloc(65536 * 2);
    _Float16* wnp   = (_Float16*)alloc(20480 * 2);
    _Float16* wep   = (_Float16*)alloc(16384 * 2);
    _Float16* featH = (_Float16*)alloc((size_t)ROWS * KPAD * 2);
    _Float16* h1H   = (_Float16*)alloc((size_t)ROWS * KPAD * 2);
    _Float16* h2H   = (_Float16*)alloc((size_t)ROWS * KPAD * 2);
    _Float16* harmH = (_Float16*)alloc((size_t)ROWS * KPAD * 2);
    float* harmCtrl = (float*)alloc((size_t)ROWS * 256 * 4);
    float* srcParam = (float*)alloc((size_t)ROWS * 256 * 4);
    float* noiParam = (float*)alloc((size_t)ROWS * 80 * 4);
    float* eF       = (float*)alloc((size_t)ROWS * 64 * 4);
    float* vibRate  = (float*)alloc(ROWS * 4);
    float* vibDepth = (float*)alloc(ROWS * 4);
    float* breath   = (float*)alloc(ROWS * 4);
    float* f0vib    = (float*)alloc(ROWS * 4);
    float* f0up     = (float*)alloc((size_t)Bsz * Ln * 4);
    float* phaseP   = (float*)alloc((size_t)Bsz * Ln * 4);
    float* chunkSum = (float*)alloc(Bsz * NCHNK * 4);
    float* chunkOff = (float*)alloc(Bsz * NCHNK * 4);
    float* harmRaw  = (float*)alloc((size_t)Bsz * Ln * 4);
    float* noiseF   = (float*)alloc((size_t)Bsz * Ln * 4);

    // 0.4 / H_80
    double h80 = 0.0;
    for (int k = 1; k <= Kh; ++k) h80 += 1.0 / (double)k;
    float coef = (float)(0.4 / h80);

    // 1. pack weights into WMMA B fragments
    svs_pack_w<<<256, 256, 0, stream>>>(W1, w1p, INDIM, 256, 65536);
    svs_pack_w<<<256, 256, 0, stream>>>(W2, w2p, 256, 256, 65536);
    svs_pack_w<<<256, 256, 0, stream>>>(Wh, whp, 256, 256, 65536);
    svs_pack_w<<<80,  256, 0, stream>>>(Wn, wnp, 256, 80, 20480);
    svs_pack_w<<<64,  256, 0, stream>>>(We, wep, 256, 64, 16384);
    // 2. features
    svs_build_feat<<<4096, 256, 0, stream>>>(mel, f0, seq, sid, lid, ptab, stab, ltab, featH);
    // 3-5. MLP via WMMA
    svs_wmma_gemm<<<512, 256, 0, stream>>>(featH, w1p, b1, nullptr, h1H, 256, 16, 1);   // gelu
    svs_wmma_gemm<<<512, 256, 0, stream>>>(h1H, w2p, b2, nullptr, h2H, 256, 16, 1);     // gelu
    svs_wmma_gemm<<<512, 256, 0, stream>>>(h2H, whp, bh, harmCtrl, harmH, 256, 16, 0);  // raw
    svs_scalefn<<<4096, 256, 0, stream>>>(harmCtrl, srcParam, ROWS * 256);
    svs_wmma_gemm<<<160, 256, 0, stream>>>(h2H, wnp, bn, noiParam, nullptr, 80, 5, 2);  // scale_fn
    svs_wmma_gemm<<<128, 256, 0, stream>>>(harmH, wep, be, eF, nullptr, 64, 4, 1);      // gelu
    // 6. heads + f0 path
    svs_heads<<<16, 256, 0, stream>>>(eF, Wvr, bvr, Wvd, bvd, Wbr, bbr,
                                      vibRate, vibDepth, breath);
    svs_f0vib<<<16, 256, 0, stream>>>(f0, vibRate, vibDepth, f0vib);
    svs_upsample<<<3840, 256, 0, stream>>>(f0vib, f0up);
    // 7. phase cumsum (3-stage scan), harmonic oscillator
    svs_scan1<<<Bsz * NCHNK, 256, 0, stream>>>(f0up, phaseP, chunkSum);
    svs_scan2<<<Bsz, 128, 0, stream>>>(chunkSum, chunkOff);
    svs_harm<<<3840, 256, 0, stream>>>(f0up, phaseP, chunkOff, harmRaw, coef);
    // 8. FFT filtering (harmonic -> out, noise -> ws), final mix
    svs_fft_filter<<<ROWS, 256, 0, stream>>>(harmRaw, srcParam, NMAGH, out);
    svs_fft_filter<<<ROWS, 256, 0, stream>>>(noise, noiParam, NMAGN, noiseF);
    svs_mix<<<3840, 256, 0, stream>>>(out, noiseF, breath);
}
